// graph_block_43447889166846
// MI455X (gfx1250) — compile-verified, matrix operand order checked by reading, not run on hardware
//
#include <hip/hip_runtime.h>
#include <math.h>

#define N_NODES 2048
#define C_DIM   512
#define E_EDGES 8192
#define EPS     1e-5f
#define MASK_FILL -10000.0f
#define NC (N_NODES * C_DIM)
#define EC (E_EDGES * C_DIM)

typedef __attribute__((ext_vector_type(16))) __bf16 v16bf;
typedef __attribute__((ext_vector_type(8)))  float  v8f;
typedef __attribute__((ext_vector_type(4)))  int    v4i;

struct U2 { unsigned int x, y; };
struct U4 { unsigned int x, y, z, w; };
struct U8 { U4 a, b; };                 // 32 bytes == v16bf
struct F4 { float x, y, z, w; };

#define LDS_STRIDE 40                   // ushorts/row (32 data + 8 pad): 16B-aligned runs, no bank conflicts

#if defined(__has_builtin)
#  if __has_builtin(__builtin_amdgcn_global_load_async_to_lds_b128) && \
      __has_builtin(__builtin_amdgcn_s_wait_asynccnt)
#    define HAVE_ASYNC_LDS 1
#  endif
#endif
#ifndef HAVE_ASYNC_LDS
#  define HAVE_ASYNC_LDS 0
#endif

#define GAS __attribute__((address_space(1)))
#define LAS __attribute__((address_space(3)))

__device__ __forceinline__ unsigned short f2bf(float f) {
    unsigned int u = __builtin_bit_cast(unsigned int, f);
    u += 0x7fffu + ((u >> 16) & 1u);    // RNE
    return (unsigned short)(u >> 16);
}
__device__ __forceinline__ float bf2f_lo(unsigned int u) { return __builtin_bit_cast(float, u << 16); }
__device__ __forceinline__ float bf2f_hi(unsigned int u) { return __builtin_bit_cast(float, u & 0xffff0000u); }

// 16B global -> LDS copy; async DMA (ASYNCcnt) when the toolchain exposes it.
__device__ __forceinline__ void copy16_g2l(const unsigned short* g, unsigned short* l) {
#if HAVE_ASYNC_LDS
    __builtin_amdgcn_global_load_async_to_lds_b128((GAS v4i*)g, (LAS v4i*)l, 0, 0);
#else
    *(U4*)l = *(const U4*)g;
#endif
}
__device__ __forceinline__ void wait_async_prev() {
#if HAVE_ASYNC_LDS
    __builtin_amdgcn_s_wait_asynccnt(3);   // our wave's previous stage (3 ops) complete
#endif
}
__device__ __forceinline__ void wait_async_all() {
#if HAVE_ASYNC_LDS
    __builtin_amdgcn_s_wait_asynccnt(0);
#endif
}

__device__ __forceinline__ v16bf load_frag(const unsigned short* lds, int row, int half) {
    // ISA 7.12.2 16-bit operand layout: lane<16 holds K {0..7,16..23}, lane>=16 holds K {8..15,24..31}
    const U4* p0 = (const U4*)(lds + row * LDS_STRIDE + half * 8);
    const U4* p1 = (const U4*)(lds + row * LDS_STRIDE + 16 + half * 8);
    U8 t; t.a = *p0; t.b = *p1;
    return __builtin_bit_cast(v16bf, t);
}

// OUT[m,n] = sum_k A[m,k] * W[n*wstride + wcoff + k] (+bias[n]) (+res[m,n])
// A, W bf16; f32 accumulate. Block tile 64(M)x128(N), 8 waves, wave tile 16x64.
// Double-buffered LDS staging via async-to-LDS prefetch.
template <bool HB, bool HR, bool OF, bool OB>
__global__ __launch_bounds__(256) void gemm_bf16_wmma(
    const unsigned short* __restrict__ A, const unsigned short* __restrict__ W,
    int wstride, int wcoff,
    const float* __restrict__ bias, const float* __restrict__ res,
    float* __restrict__ out, unsigned short* __restrict__ out_bf)
{
    __shared__ __align__(16) unsigned short sA[2][64 * LDS_STRIDE];
    __shared__ __align__(16) unsigned short sW[2][128 * LDS_STRIDE];

    const int t    = threadIdx.x;
    const int lane = t & 31;
    const int wave = t >> 5;
    const int wm   = wave >> 1;          // 0..3
    const int wn   = wave & 1;           // 0..1
    const int r    = lane & 15;
    const int half = lane >> 4;
    const int m0   = blockIdx.y * 64;
    const int n0   = blockIdx.x * 128;

    const int arow = t >> 2, aq = t & 3; // A tile: 64 rows x 32 bf16, 16B/thread
    const int wrow = t >> 1, wh = t & 1; // W tile: 128 rows x 32 bf16, 32B/thread

    const unsigned short* aSrc = A + (size_t)(m0 + arow) * C_DIM + aq * 8;
    const unsigned short* wSrc = W + (size_t)(n0 + wrow) * wstride + wcoff + wh * 16;
    unsigned short* aDst0 = &sA[0][arow * LDS_STRIDE + aq * 8];
    unsigned short* wDst0 = &sW[0][wrow * LDS_STRIDE + wh * 16];
    unsigned short* aDst1 = &sA[1][arow * LDS_STRIDE + aq * 8];
    unsigned short* wDst1 = &sW[1][wrow * LDS_STRIDE + wh * 16];

    v8f acc[4] = {};

    // prologue: stage K-slice 0 into buffer 0
    copy16_g2l(aSrc, aDst0);
    copy16_g2l(wSrc, wDst0);
    copy16_g2l(wSrc + 8, wDst0 + 8);

    const int KT = C_DIM / 32;
    for (int kt = 0; kt < KT; ++kt) {
        const int buf = kt & 1;
        if (kt + 1 < KT) {               // prefetch next slice into the other buffer
            const int kk = (kt + 1) * 32;
            unsigned short* ad = buf ? aDst0 : aDst1;
            unsigned short* wd = buf ? wDst0 : wDst1;
            copy16_g2l(aSrc + kk, ad);
            copy16_g2l(wSrc + kk, wd);
            copy16_g2l(wSrc + kk + 8, wd + 8);
            wait_async_prev();           // slice kt resident (in-order completion)
        } else {
            wait_async_all();
        }
        __syncthreads();

        v16bf a = load_frag(sA[buf], wm * 16 + r, half);
#pragma unroll
        for (int nb = 0; nb < 4; ++nb) {
            v16bf b = load_frag(sW[buf], wn * 64 + nb * 16 + r, half);
            acc[nb] = __builtin_amdgcn_wmma_f32_16x16x32_bf16(
                false, a, false, b, (short)0, acc[nb], false, false);
        }
        __syncthreads();                 // buffer safe to overwrite next iteration
    }

#pragma unroll
    for (int nb = 0; nb < 4; ++nb) {
        const int col = n0 + wn * 64 + nb * 16 + r;
        const float bv = HB ? bias[col] : 0.0f;
#pragma unroll
        for (int i = 0; i < 8; ++i) {
            const int rowg = m0 + wm * 16 + half * 8 + i;  // C/D layout: vgpr i -> M=i (+8 upper half)
            const size_t idx = (size_t)rowg * C_DIM + col;
            float val = acc[nb][i] + bv;
            if (HR) val += res[idx];
            if (OF) out[idx] = val;
            if (OB) out_bf[idx] = f2bf(val);
        }
    }
}

// f32 -> bf16 bulk convert, 4 elems/thread
__global__ __launch_bounds__(256) void cvt_bf16_kernel(
    const float* __restrict__ in, unsigned short* __restrict__ out)
{
    const int i = blockIdx.x * 256 + threadIdx.x;
    F4 v = ((const F4*)in)[i];
    U2 p;
    p.x = (unsigned int)f2bf(v.x) | ((unsigned int)f2bf(v.y) << 16);
    p.y = (unsigned int)f2bf(v.z) | ((unsigned int)f2bf(v.w) << 16);
    ((U2*)out)[i] = p;
}

// rootT_bf[n,k] = bf16(root[k,n])
__global__ __launch_bounds__(256) void transpose512_bf(
    const float* __restrict__ in, unsigned short* __restrict__ out)
{
    __shared__ float tile[16][17];
    int x = blockIdx.x * 16 + threadIdx.x;
    int y = blockIdx.y * 16 + threadIdx.y;
    tile[threadIdx.y][threadIdx.x] = in[(size_t)y * 512 + x];
    __syncthreads();
    out[(size_t)(blockIdx.x * 16 + threadIdx.y) * 512 + blockIdx.y * 16 + threadIdx.x] =
        f2bf(tile[threadIdx.x][threadIdx.y]);
}

// s[e] = leaky_relu( dot(tanh(Ai[dst]+Aj[src]+Ae[e]+b1), w2), 0.2 ), one wave per edge
__global__ __launch_bounds__(256) void score_kernel(
    const float* __restrict__ Ai, const float* __restrict__ Aj, const float* __restrict__ Ae,
    const float* __restrict__ b1, const float* __restrict__ w2,
    const long long* __restrict__ ei, float* __restrict__ s)
{
    const int lane = threadIdx.x & 31;
    const int e = blockIdx.x * 8 + (threadIdx.x >> 5);
    const long long sn = ei[e];
    const long long dn = ei[E_EDGES + e];
    const float* ai = Ai + (size_t)dn * C_DIM;
    const float* aj = Aj + (size_t)sn * C_DIM;
    const float* ae = Ae + (size_t)e  * C_DIM;
    float acc = 0.0f;
#pragma unroll
    for (int i = 0; i < 16; ++i) {
        int c = i * 32 + lane;
        acc += tanhf(ai[c] + aj[c] + ae[c] + b1[c]) * w2[c];
    }
#pragma unroll
    for (int off = 16; off > 0; off >>= 1) acc += __shfl_xor(acc, off, 32);
    if (lane == 0) s[e] = acc > 0.0f ? acc : 0.2f * acc;
}

// lse[row] = logsumexp_k (mask[row,k] ? s[k] : -1e4). One block/row, single NT pass over 64MB mask.
__global__ __launch_bounds__(256) void lse_kernel(
    const unsigned char* __restrict__ mask, const float* __restrict__ s, float* __restrict__ lse)
{
    const int row = blockIdx.x, t = threadIdx.x;
    const unsigned int* mrow = (const unsigned int*)(mask + (size_t)row * E_EDGES);
    float m = -3.0e38f, sum = 0.0f;
#pragma unroll
    for (int j = 0; j < 8; ++j) {
        unsigned int w = __builtin_nontemporal_load(mrow + t * 8 + j);
#pragma unroll
        for (int b = 0; b < 4; ++b) {
            int col = t * 32 + j * 4 + b;
            float l = ((w >> (8 * b)) & 0xffu) ? s[col] : MASK_FILL;
            if (l > m) { sum = sum * __expf(m - l) + 1.0f; m = l; }
            else       { sum += __expf(l - m); }
        }
    }
    __shared__ float sm[256], ss[256];
    sm[t] = m; ss[t] = sum;
    __syncthreads();
    for (int off = 128; off > 0; off >>= 1) {
        if (t < off) {
            float m2 = sm[t + off], s2 = ss[t + off];
            float M = fmaxf(sm[t], m2);
            ss[t] = ss[t] * __expf(sm[t] - M) + s2 * __expf(m2 - M);
            sm[t] = M;
        }
        __syncthreads();
    }
    if (t == 0) lse[row] = sm[0] + __logf(ss[0]);
}

__global__ __launch_bounds__(256) void att_kernel(
    const unsigned char* __restrict__ mask, const float* __restrict__ s,
    const float* __restrict__ lse, float* __restrict__ att)
{
    int e = blockIdx.x * 256 + threadIdx.x;
    float d = mask[(size_t)e * (E_EDGES + 1)] ? s[e] : MASK_FILL;
    att[e] = __expf(d - lse[e]);
}

// O[dst[e], :] += bf2f(Hbf[src[e], :]) * att[e]   (4 edges/block, 8 channels/thread)
__global__ __launch_bounds__(256) void aggregate_kernel(
    const unsigned short* __restrict__ Hbf, const float* __restrict__ att,
    const long long* __restrict__ ei, float* __restrict__ O)
{
    const int t = threadIdx.x;
    const int e = blockIdx.x * 4 + (t >> 6);
    const int c = (t & 63) * 8;
    const float a = att[e];
    const long long sn = ei[e];
    const long long dn = ei[E_EDGES + e];
    U4 raw = *(const U4*)(Hbf + (size_t)sn * C_DIM + c);
    float* o = O + (size_t)dn * C_DIM + c;
    unsigned int u[4] = { raw.x, raw.y, raw.z, raw.w };
#pragma unroll
    for (int i = 0; i < 4; ++i) {
        __hip_atomic_fetch_add(o + 2*i,     bf2f_lo(u[i]) * a, __ATOMIC_RELAXED, __HIP_MEMORY_SCOPE_AGENT);
        __hip_atomic_fetch_add(o + 2*i + 1, bf2f_hi(u[i]) * a, __ATOMIC_RELAXED, __HIP_MEMORY_SCOPE_AGENT);
    }
}

__global__ __launch_bounds__(256) void ln_kernel(
    const float* __restrict__ in, const float* __restrict__ g, const float* __restrict__ b,
    float* __restrict__ out, unsigned short* __restrict__ out_bf)
{
    const int row = blockIdx.x, t = threadIdx.x;
    const float* r = in + (size_t)row * C_DIM;
    float v0 = r[t], v1 = r[t + 256];
    __shared__ float s1[256], s2[256];
    s1[t] = v0 + v1; s2[t] = v0 * v0 + v1 * v1;
    __syncthreads();
    for (int off = 128; off > 0; off >>= 1) {
        if (t < off) { s1[t] += s1[t + off]; s2[t] += s2[t + off]; }
        __syncthreads();
    }
    float m   = s1[0] * (1.0f / C_DIM);
    float var = s2[0] * (1.0f / C_DIM) - m * m;
    float iv  = rsqrtf(var + EPS);
    float o0 = (v0 - m) * iv * g[t] + b[t];
    float o1 = (v1 - m) * iv * g[t + 256] + b[t + 256];
    out[(size_t)row * C_DIM + t]       = o0;
    out[(size_t)row * C_DIM + t + 256] = o1;
    if (out_bf) {
        out_bf[(size_t)row * C_DIM + t]       = f2bf(o0);
        out_bf[(size_t)row * C_DIM + t + 256] = f2bf(o1);
    }
}

extern "C" void kernel_launch(void* const* d_in, const int* in_sizes, int n_in,
                              void* d_out, int out_size, void* d_ws, size_t ws_size,
                              hipStream_t stream) {
    (void)in_sizes; (void)n_in; (void)out_size; (void)ws_size;
    const float*          x      = (const float*)d_in[0];
    const long long*      ei     = (const long long*)d_in[1];
    const float*          eattr  = (const float*)d_in[2];
    const unsigned char*  masks  = (const unsigned char*)d_in[3];
    const float*          W_nn   = (const float*)d_in[4];
    const float*          b_nn   = (const float*)d_in[5];
    const float*          W_att1 = (const float*)d_in[6];
    const float*          b_att1 = (const float*)d_in[7];
    const float*          w_att2 = (const float*)d_in[8];
    const float*          root   = (const float*)d_in[9];
    const float*          bias   = (const float*)d_in[10];
    const float*          g1     = (const float*)d_in[11];
    const float*          be1    = (const float*)d_in[12];
    const float*          g2     = (const float*)d_in[13];
    const float*          be2    = (const float*)d_in[14];
    const float*          W_lin  = (const float*)d_in[15];
    const float*          b_lin  = (const float*)d_in[16];
    float* outp = (float*)d_out;

    // ---- workspace layout ----
    float* ws  = (float*)d_ws;
    float* Ai  = ws;                        // [N,C] f32
    float* Aj  = Ai + NC;                   // [N,C]
    float* Ae  = Aj + NC;                   // [E,C]
    float* O   = Ae + EC;                   // [N,C]
    float* x1  = O + NC;                    // [N,C]
    float* y   = x1 + NC;                   // [N,C]
    float* s   = y + NC;                    // [E]
    float* lse = s + E_EDGES;               // [E]
    float* att = lse + E_EDGES;             // [E]
    unsigned short* x_bf    = (unsigned short*)(att + E_EDGES);
    unsigned short* ea_bf   = x_bf   + NC;                  // [E,C]
    unsigned short* H_bf    = ea_bf  + EC;                  // [N,C]
    unsigned short* He_bf   = H_bf   + NC;                  // [E,C]
    unsigned short* x1_bf   = He_bf  + EC;                  // [N,C]
    unsigned short* Wnn_bf  = x1_bf  + NC;                  // [C,C]
    unsigned short* Wat_bf  = Wnn_bf + C_DIM * C_DIM;       // [C,3C]
    unsigned short* rootT_bf= Wat_bf + (size_t)C_DIM * 3 * C_DIM; // [C,C]
    unsigned short* Wlin_bf = rootT_bf + C_DIM * C_DIM;     // [C,C]

    const dim3 blk(256);
    const dim3 gN(4, N_NODES / 64);
    const dim3 gE(4, E_EDGES / 64);

    // one-shot bf16 conversions
    cvt_bf16_kernel<<<NC / 1024, blk, 0, stream>>>(x, x_bf);
    cvt_bf16_kernel<<<EC / 1024, blk, 0, stream>>>(eattr, ea_bf);
    cvt_bf16_kernel<<<C_DIM * C_DIM / 1024, blk, 0, stream>>>(W_nn, Wnn_bf);
    cvt_bf16_kernel<<<C_DIM * 3 * C_DIM / 1024, blk, 0, stream>>>(W_att1, Wat_bf);
    cvt_bf16_kernel<<<C_DIM * C_DIM / 1024, blk, 0, stream>>>(W_lin, Wlin_bf);
    transpose512_bf<<<dim3(32, 32), dim3(16, 16), 0, stream>>>(root, rootT_bf);

    // H = x@Wnn^T+b ; He = edge_attr@Wnn^T+b  (bf16 outputs, feed later GEMMs + aggregate)
    gemm_bf16_wmma<true,  false, false, true ><<<gN, blk, 0, stream>>>(x_bf,  Wnn_bf, C_DIM, 0, b_nn, nullptr, nullptr, H_bf);
    gemm_bf16_wmma<true,  false, false, true ><<<gE, blk, 0, stream>>>(ea_bf, Wnn_bf, C_DIM, 0, b_nn, nullptr, nullptr, He_bf);
    // attention split
    gemm_bf16_wmma<false, false, true,  false><<<gN, blk, 0, stream>>>(H_bf,  Wat_bf, 3*C_DIM,       0, nullptr, nullptr, Ai, nullptr);
    gemm_bf16_wmma<false, false, true,  false><<<gN, blk, 0, stream>>>(H_bf,  Wat_bf, 3*C_DIM,   C_DIM, nullptr, nullptr, Aj, nullptr);
    gemm_bf16_wmma<false, false, true,  false><<<gE, blk, 0, stream>>>(He_bf, Wat_bf, 3*C_DIM, 2*C_DIM, nullptr, nullptr, Ae, nullptr);
    // O = x @ root + bias  (via rootT)
    gemm_bf16_wmma<true,  false, true,  false><<<gN, blk, 0, stream>>>(x_bf, rootT_bf, C_DIM, 0, bias, nullptr, O, nullptr);

    score_kernel<<<E_EDGES / 8, blk, 0, stream>>>(Ai, Aj, Ae, b_att1, w_att2, ei, s);
    lse_kernel<<<E_EDGES, blk, 0, stream>>>(masks, s, lse);
    att_kernel<<<E_EDGES / 256, blk, 0, stream>>>(masks, s, lse, att);
    aggregate_kernel<<<E_EDGES / 4, blk, 0, stream>>>(H_bf, att, ei, O);

    ln_kernel<<<N_NODES, blk, 0, stream>>>(O, g1, be1, x1, x1_bf);
    gemm_bf16_wmma<true, true, true, false><<<gN, blk, 0, stream>>>(x1_bf, Wlin_bf, C_DIM, 0, b_lin, x1, y, nullptr);
    ln_kernel<<<N_NODES, blk, 0, stream>>>(y, g2, be2, outp, nullptr);
}
